// AlltagCtxGenerator_69801808495266
// MI455X (gfx1250) — compile-verified, matrix operand order checked by reading, not run on hardware
//
#include <hip/hip_runtime.h>

// ---------------- problem constants ----------------
constexpr int BS = 16, LS = 128, HS = 768, PEXP = 8, MSUB = 1024, DEMB = 512;
constexpr int TTOK = BS * LS;          // 2048 tokens
constexpr int NPRIV = 4;
constexpr float TEMP = 1.0f;
constexpr int KB1 = HS / 32;           // 24 K-blocks in logits GEMM
constexpr int KB2 = MSUB / 32;         // 32 K-blocks in embed GEMM

// d_out layout (flat float32, reference tuple order)
constexpr size_t OFF_WORD = 0;
constexpr size_t OFF_PSR  = OFF_WORD + TTOK;
constexpr size_t OFF_ATK  = OFF_PSR + (size_t)TTOK * DEMB;
constexpr size_t OFF_ENT  = OFF_ATK + (size_t)TTOK * DEMB;
constexpr size_t OFF_CPY  = OFF_ENT + 1;
constexpr size_t OFF_OBF  = OFF_CPY + TTOK;
constexpr size_t OFF_PRI  = OFF_OBF + TTOK;

// ---------------- vector types ----------------
typedef __attribute__((ext_vector_type(16))) __bf16 v16bf;
typedef __attribute__((ext_vector_type(8)))  __bf16 v8bf;
typedef __attribute__((ext_vector_type(8)))  float  v8f;
typedef __attribute__((ext_vector_type(4)))  float  v4f;
typedef __attribute__((ext_vector_type(4)))  unsigned short v4u;

// fp32 -> bf16, round-to-nearest-even
static __device__ __forceinline__ unsigned short f2bf(float f) {
    unsigned int u = __builtin_bit_cast(unsigned int, f);
    u += 0x7FFFu + ((u >> 16) & 1u);
    return (unsigned short)(u >> 16);
}
static __device__ __forceinline__ float wred_max(float v) {
    #pragma unroll
    for (int o = 16; o > 0; o >>= 1) v = fmaxf(v, __shfl_xor(v, o, 32));
    return v;
}
static __device__ __forceinline__ float wred_sum(float v) {
    #pragma unroll
    for (int o = 16; o > 0; o >>= 1) v += __shfl_xor(v, o, 32);
    return v;
}
// build a 16x32 fragment register from the two contiguous 8-element runs
static __device__ __forceinline__ v16bf frag16(const unsigned short* base, int g) {
    v8bf lo = *(const v8bf*)(base + g * 8);        // K = g*8 .. g*8+7
    v8bf hi = *(const v8bf*)(base + 16 + g * 8);   // K = 16+g*8 .. 16+g*8+7
    return __builtin_shufflevector(lo, hi, 0,1,2,3,4,5,6,7,8,9,10,11,12,13,14,15);
}

// =====================================================================
// P0: elementwise fp32 -> bf16 (4 elems / thread), used for ctx
// =====================================================================
__global__ __launch_bounds__(256) void k_cvt_bf16(
    const float* __restrict__ src, unsigned short* __restrict__ dst, int n4)
{
    int i = blockIdx.x * 256 + threadIdx.x;
    if (i < n4) {
        v4f v = ((const v4f*)src)[i];
        v4u o;
        o[0] = f2bf(v[0]); o[1] = f2bf(v[1]); o[2] = f2bf(v[2]); o[3] = f2bf(v[3]);
        ((v4u*)dst)[i] = o;
    }
}

// =====================================================================
// P1: dec_W [P][HS][M] fp32 -> bf16 swizzled [p][k/32][n][k%32]
//     block handles one (p, kb, 64-col n chunk); transpose via LDS
// =====================================================================
__global__ __launch_bounds__(256) void k_swz_decw(
    const float* __restrict__ decW, unsigned short* __restrict__ dst)
{
    const int n0 = blockIdx.x * 64;
    const int kb = blockIdx.y;
    const int p  = blockIdx.z;
    __shared__ float t[32][65];
    const int tid = threadIdx.x;
    #pragma unroll
    for (int j = 0; j < 8; ++j) {                 // 2048 elems, 8/thread
        int idx = tid + j * 256;
        int r = idx >> 6, c = idx & 63;
        t[r][c] = decW[(size_t)p * HS * MSUB + (size_t)(kb * 32 + r) * MSUB + n0 + c];
    }
    __syncthreads();
    #pragma unroll
    for (int j = 0; j < 8; ++j) {
        int idx = tid + j * 256;
        int n = idx >> 5, kk = idx & 31;
        dst[(((size_t)(p * KB1 + kb) * MSUB + n0 + n) << 5) + kk] = f2bf(t[kk][n]);
    }
}

// =====================================================================
// P2: gathered embedding LUT, bf16 swizzled [p][m/32][n][m%32]
//     lut[p][m][d] = weight[words[p][m]][d];  block = (p, mb, 128-col chunk)
// =====================================================================
__global__ __launch_bounds__(256) void k_build_lut(
    const float* __restrict__ W, const int* __restrict__ words,
    unsigned short* __restrict__ dst)
{
    const int n0 = blockIdx.x * 128;
    const int mb = blockIdx.y;
    const int p  = blockIdx.z;
    __shared__ float t[32][129];
    __shared__ int sRow[32];
    const int tid = threadIdx.x;
    if (tid < 32) sRow[tid] = words[p * MSUB + mb * 32 + tid];
    __syncthreads();
    #pragma unroll
    for (int j = 0; j < 16; ++j) {                // 4096 elems, 16/thread
        int idx = tid + j * 256;
        int r = idx >> 7, c = idx & 127;
        t[r][c] = W[(size_t)sRow[r] * DEMB + n0 + c];
    }
    __syncthreads();
    #pragma unroll
    for (int j = 0; j < 16; ++j) {
        int idx = tid + j * 256;
        int n = idx >> 5, kk = idx & 31;
        dst[(((size_t)(p * KB2 + mb) * DEMB + n0 + n) << 5) + kk] = f2bf(t[kk][n]);
    }
}

// =====================================================================
// K1: per (16-token tile, expert p): logits GEMM (bf16 WMMA, fp32 acc),
//     log-softmax, entropy term, gumbel-softmax, argmax word, spt(bf16)->ws
// =====================================================================
__global__ __launch_bounds__(256) void k_expert_logits(
    const int* __restrict__ pos, const unsigned short* __restrict__ ctx_bf,
    const unsigned short* __restrict__ decw_swz, const float* __restrict__ decb,
    const float* __restrict__ ugum, const int* __restrict__ words,
    unsigned short* __restrict__ spt_bf, float* __restrict__ negent_ws,
    float* __restrict__ out)
{
    const int p    = blockIdx.y;
    const int t0   = blockIdx.x * 16;
    const int tid  = threadIdx.x;
    const int lane = tid & 31;
    const int wv   = tid >> 5;

    __shared__ int   s_pos[16];
    __shared__ int   s_any;
    __shared__ float sL[16 * MSUB];   // logits tile, fp32

    if (tid == 0) s_any = 0;
    __syncthreads();
    if (tid < 16) {
        int pv = pos[t0 + tid];
        s_pos[tid] = pv;
        if (pv == p) s_any = 1;
    }
    __syncthreads();
    if (!s_any) return;

    const int g  = lane >> 4;
    const int rn = lane & 15;
    const unsigned short* Arow = ctx_bf + (size_t)(t0 + rn) * HS;

    v8f acc[8];
    #pragma unroll
    for (int s = 0; s < 8; ++s)
        #pragma unroll
        for (int r = 0; r < 8; ++r) acc[s][r] = 0.0f;

    for (int kbi = 0; kbi < KB1; ++kbi) {
        const v16bf a = frag16(Arow + kbi * 32, g);
        const unsigned short* Bk = decw_swz + ((size_t)(p * KB1 + kbi) * MSUB << 5);
        #pragma unroll
        for (int s = 0; s < 8; ++s) {
            const int n = wv * 128 + s * 16 + rn;
            const v16bf b = frag16(Bk + ((size_t)n << 5), g);
            acc[s] = __builtin_amdgcn_wmma_f32_16x16x32_bf16(
                false, a, false, b, (short)0, acc[s], false, false);
        }
    }

    // logits + bias -> LDS  (C layout: VGPR r -> row r + 8*g, col rn)
    #pragma unroll
    for (int s = 0; s < 8; ++s) {
        const int col = wv * 128 + s * 16 + rn;
        const float bias = decb[p * MSUB + col];
        #pragma unroll
        for (int r = 0; r < 8; ++r)
            sL[(g * 8 + r) * MSUB + col] = acc[s][r] + bias;
    }
    __syncthreads();

    // each wave post-processes 2 tokens
    for (int q = 0; q < 2; ++q) {
        const int tt = wv * 2 + q;
        if (s_pos[tt] != p) continue;
        const int tg = t0 + tt;
        const float* row = &sL[tt * MSUB];

        float m1 = -3.4e38f;
        #pragma unroll
        for (int i = 0; i < 32; ++i) m1 = fmaxf(m1, row[lane + (i << 5)]);
        m1 = wred_max(m1);

        float z = 0.0f;
        #pragma unroll
        for (int i = 0; i < 32; ++i) z += expf(row[lane + (i << 5)] - m1);
        z = wred_sum(z);
        const float lse = m1 + logf(z);

        float ne = 0.0f;
        #pragma unroll
        for (int i = 0; i < 32; ++i) {
            float l = row[lane + (i << 5)] - lse;   // pspt
            ne -= l * expf(l);
        }
        ne = wred_sum(ne);

        // gumbel softmax
        float sl[32];
        float mx2 = -3.4e38f; int bi = 0;
        #pragma unroll
        for (int i = 0; i < 32; ++i) {
            const int c = lane + (i << 5);
            float u = ugum[(size_t)tg * MSUB + c];
            u = fminf(fmaxf(u, 1e-6f), 1.0f - 1e-6f);
            const float gg = -logf(-logf(u));
            const float v  = (row[c] - lse + gg) / TEMP;
            sl[i] = v;
            if (v > mx2) { mx2 = v; bi = c; }
        }
        #pragma unroll
        for (int o = 16; o > 0; o >>= 1) {
            float ov = __shfl_xor(mx2, o, 32);
            int   oi = __shfl_xor(bi, o, 32);
            if (ov > mx2 || (ov == mx2 && oi < bi)) { mx2 = ov; bi = oi; }
        }
        float z2 = 0.0f;
        #pragma unroll
        for (int i = 0; i < 32; ++i) z2 += expf(sl[i] - mx2);
        z2 = wred_sum(z2);
        const float inv = 1.0f / z2;
        #pragma unroll
        for (int i = 0; i < 32; ++i)
            spt_bf[(size_t)tg * MSUB + lane + (i << 5)] = f2bf(expf(sl[i] - mx2) * inv);

        if (lane == 0) {
            negent_ws[tg] = ne;
            out[OFF_WORD + tg] = (float)words[p * MSUB + bi];
        }
    }
}

// =====================================================================
// K2: per (tile, expert p): psr/atk embedding mix GEMM via WMMA.
//     A = spt tile (bf16, ws); B = pre-gathered swizzled LUT (bf16, ws).
//     waves 0..3 -> psr (4*128 = 512 cols), waves 4..7 -> atk
// =====================================================================
__global__ __launch_bounds__(256) void k_expert_embed(
    const int* __restrict__ pos, const unsigned short* __restrict__ spt_bf,
    const unsigned short* __restrict__ psr_swz,
    const unsigned short* __restrict__ atk_swz, float* __restrict__ out)
{
    const int p    = blockIdx.y;
    const int t0   = blockIdx.x * 16;
    const int tid  = threadIdx.x;
    const int lane = tid & 31;
    const int wv   = tid >> 5;

    __shared__ int s_pos[16];
    __shared__ int s_any;

    if (tid == 0) s_any = 0;
    __syncthreads();
    if (tid < 16) {
        int pv = pos[t0 + tid];
        s_pos[tid] = pv;
        if (pv == p) s_any = 1;
    }
    __syncthreads();
    if (!s_any) return;

    const bool is_atk = wv >= 4;
    const unsigned short* Wsw = is_atk ? atk_swz : psr_swz;
    const size_t out_base = is_atk ? OFF_ATK : OFF_PSR;
    const int wn = (wv & 3) * 128;

    const int g  = lane >> 4;
    const int rn = lane & 15;
    const unsigned short* Arow = spt_bf + (size_t)(t0 + rn) * MSUB;

    v8f acc[8];
    #pragma unroll
    for (int s = 0; s < 8; ++s)
        #pragma unroll
        for (int r = 0; r < 8; ++r) acc[s][r] = 0.0f;

    for (int kbi = 0; kbi < KB2; ++kbi) {
        const v16bf a = frag16(Arow + kbi * 32, g);
        const unsigned short* Bk = Wsw + ((size_t)(p * KB2 + kbi) * DEMB << 5);
        #pragma unroll
        for (int s = 0; s < 8; ++s) {
            const int n = wn + s * 16 + rn;
            const v16bf b = frag16(Bk + ((size_t)n << 5), g);
            acc[s] = __builtin_amdgcn_wmma_f32_16x16x32_bf16(
                false, a, false, b, (short)0, acc[s], false, false);
        }
    }

    #pragma unroll
    for (int s = 0; s < 8; ++s) {
        const int col = wn + s * 16 + rn;
        #pragma unroll
        for (int r = 0; r < 8; ++r) {
            const int rowi = g * 8 + r;
            if (s_pos[rowi] == p)
                out[out_base + (size_t)(t0 + rowi) * DEMB + col] = acc[s][r];
        }
    }
}

// =====================================================================
// K3: passthrough words, masks, deterministic entropy reduction
// =====================================================================
__global__ __launch_bounds__(256) void k_finalize(
    const int* __restrict__ pos, const int* __restrict__ word,
    const int* __restrict__ mask, const float* __restrict__ negent_ws,
    float* __restrict__ out)
{
    const int t = blockIdx.x * 256 + threadIdx.x;
    if (t < TTOK) {
        const int pv = pos[t];
        const int w  = word[t];
        float ow;
        if (pv < PEXP) {
            ow = out[OFF_WORD + t];          // written by K1 this call
        } else {
            ow = (float)w;
            out[OFF_WORD + t] = ow;
        }
        const int owi = (int)(ow + 0.5f);
        out[OFF_CPY + t] = (owi == w && mask[t] != 0) ? 1.0f : 0.0f;
        out[OFF_OBF + t] = (pv < PEXP) ? 1.0f : 0.0f;
        out[OFF_PRI + t] = (pv < NPRIV && pv < PEXP) ? 1.0f : 0.0f;
    }
    if (blockIdx.x == 0 && threadIdx.x == 0) {
        float sum[PEXP]; int cnt[PEXP];
        #pragma unroll
        for (int i = 0; i < PEXP; ++i) { sum[i] = 0.0f; cnt[i] = 0; }
        for (int tt = 0; tt < TTOK; ++tt) {
            const int pv = pos[tt];
            if (pv < PEXP) { sum[pv] += negent_ws[tt]; cnt[pv]++; }
        }
        float ent = 0.0f;
        #pragma unroll
        for (int i = 0; i < PEXP; ++i)
            if (cnt[i] > 0) ent += sum[i] / ((float)cnt[i] * (float)MSUB);
        out[OFF_ENT] = -ent;
    }
}

// =====================================================================
// K4: passthrough embeddings for non-target tokens (pos >= P)
// =====================================================================
__global__ __launch_bounds__(256) void k_passthrough(
    const int* __restrict__ pos, const int* __restrict__ word,
    const float* __restrict__ psr_w, const float* __restrict__ atk_w,
    float* __restrict__ out)
{
    const int t = blockIdx.x;
    if (pos[t] >= PEXP) {
        const size_t w = (size_t)word[t];
        for (int i = threadIdx.x; i < DEMB; i += 256) {
            out[OFF_PSR + (size_t)t * DEMB + i] = psr_w[w * DEMB + i];
            out[OFF_ATK + (size_t)t * DEMB + i] = atk_w[w * DEMB + i];
        }
    }
}

// =====================================================================
extern "C" void kernel_launch(void* const* d_in, const int* in_sizes, int n_in,
                              void* d_out, int out_size, void* d_ws, size_t ws_size,
                              hipStream_t stream) {
    (void)in_sizes; (void)n_in; (void)out_size; (void)ws_size;
    const int*   inp_word = (const int*)d_in[0];
    const int*   inp_pos  = (const int*)d_in[1];
    const int*   inp_mask = (const int*)d_in[2];
    const float* ctx      = (const float*)d_in[3];
    const float* decW     = (const float*)d_in[4];
    const float* decb     = (const float*)d_in[5];
    const float* psr_w    = (const float*)d_in[6];
    const float* atk_w    = (const float*)d_in[7];
    const int*   words    = (const int*)d_in[8];
    const float* ugum     = (const float*)d_in[9];
    float* out = (float*)d_out;

    // ---- workspace carve-up (bytes) ----
    char* wsb = (char*)d_ws;
    unsigned short* spt_bf   = (unsigned short*)wsb;                    // 4 MB
    wsb += (size_t)TTOK * MSUB * 2;
    float* negent_ws         = (float*)wsb;                             // 8 KB
    wsb += (size_t)TTOK * 4;
    unsigned short* ctx_bf   = (unsigned short*)wsb;                    // 3 MB
    wsb += (size_t)TTOK * HS * 2;
    unsigned short* decw_swz = (unsigned short*)wsb;                    // 12.6 MB
    wsb += (size_t)PEXP * HS * MSUB * 2;
    unsigned short* psr_swz  = (unsigned short*)wsb;                    // 8.4 MB
    wsb += (size_t)PEXP * MSUB * DEMB * 2;
    unsigned short* atk_swz  = (unsigned short*)wsb;                    // 8.4 MB

    // ---- pre-passes: convert / swizzle / gather into bf16 ----
    {
        int n4 = TTOK * HS / 4;
        k_cvt_bf16<<<(n4 + 255) / 256, 256, 0, stream>>>(ctx, ctx_bf, n4);
    }
    k_swz_decw<<<dim3(MSUB / 64, KB1, PEXP), 256, 0, stream>>>(decW, decw_swz);
    k_build_lut<<<dim3(DEMB / 128, KB2, PEXP), 256, 0, stream>>>(psr_w, words, psr_swz);
    k_build_lut<<<dim3(DEMB / 128, KB2, PEXP), 256, 0, stream>>>(atk_w, words, atk_swz);

    // ---- main fused pipeline ----
    dim3 grid(TTOK / 16, PEXP);
    k_expert_logits<<<grid, 256, 0, stream>>>(inp_pos, ctx_bf, decw_swz, decb,
                                              ugum, words, spt_bf, negent_ws, out);
    k_expert_embed<<<grid, 256, 0, stream>>>(inp_pos, spt_bf, psr_swz, atk_swz, out);
    k_finalize<<<(TTOK + 255) / 256, 256, 0, stream>>>(inp_pos, inp_word,
                                                       inp_mask, negent_ws, out);
    k_passthrough<<<TTOK, 256, 0, stream>>>(inp_pos, inp_word, psr_w, atk_w, out);
}
